// ExplainableHeteroClassifier_without_lexical_chain_23742579212720
// MI455X (gfx1250) — matrix operand order repaired and checked by reading, not compiled, not run on hardware
//
#include <hip/hip_runtime.h>
#include <hip/hip_bf16.h>

typedef __bf16 bf16;
typedef __attribute__((ext_vector_type(16))) __bf16 v16bf;
typedef __attribute__((ext_vector_type(8)))  __bf16 bf16x8;
typedef __attribute__((ext_vector_type(8)))  float  v8f;

#define R_   20
#define N_   4096
#define B_   16
#define S_   256
#define DIN_ 512
#define HID_ 128
#define H_   4
#define D_   512
#define E_   16384

union AFrag { v16bf v; bf16x8 h[2]; };

// Async copy of 32 contiguous bytes (global -> LDS), per lane, via the CDNA5
// Tensor/async path (ASYNCcnt). INST_OFFSET applies to both LDS and global
// addresses, so one address pair covers both 16B halves.
__device__ __forceinline__ void async_cp32B(unsigned lds, const bf16* g) {
    asm volatile("global_load_async_to_lds_b128 %0, %1, off\n\t"
                 "global_load_async_to_lds_b128 %0, %1, off offset:16"
                 :: "v"(lds), "v"(g) : "memory");
}
__device__ __forceinline__ void wait_async0() {
    asm volatile("s_wait_asynccnt 0x0" ::: "memory");
}

// ---------------------------------------------------------------------------
// WMMA GEMM: C[M,N] (f32) = act(A[M,K] @ Bt[N,K]^T + bias)
// A bf16 row-major [M,K]; Bt bf16 row-major [N,K] (pre-transposed weights).
// Block tile 128x128, BK=32, 256 threads = 8 waves, wave tile 32x64.
// Double-buffered LDS staged by GLOBAL_LOAD_ASYNC_TO_LDS_B128; one barrier
// per K-tile; async loads for tile t+1 overlap WMMA compute of tile t.
// ---------------------------------------------------------------------------
__global__ __launch_bounds__(256) void k_gemm_bf16_wmma(
    const bf16* __restrict__ A, const bf16* __restrict__ Bt, float* __restrict__ C,
    int M, int N, int K, const float* __restrict__ bias, int act)
{
    __shared__ __align__(16) bf16 sA[2][128 * 40];   // [buf][m_row][32 + 8 pad]
    __shared__ __align__(16) bf16 sB[2][128 * 40];   // [buf][n_row][32 + 8 pad]

    const int tid   = threadIdx.x;
    const int lane  = tid & 31;
    const int w     = tid >> 5;
    const int waveM = w & 3;        // 4 waves along M
    const int waveN = w >> 2;       // 2 waves along N
    const int half  = lane >> 4;
    const int l16   = lane & 15;
    const int bm    = blockIdx.y * 128;
    const int bn    = blockIdx.x * 128;

    v8f acc[2][4];
#pragma unroll
    for (int mi = 0; mi < 2; ++mi)
#pragma unroll
        for (int ni = 0; ni < 4; ++ni)
            acc[mi][ni] = (v8f){0.f,0.f,0.f,0.f,0.f,0.f,0.f,0.f};

    // staging: 2 threads per row, 32 bytes each; identical pattern for A and Bt
    const int row  = tid >> 1;
    const int seg  = (tid & 1) * 16;
    const int soff = row * 40 + seg;
    const bf16* gA = A  + (size_t)(bm + row) * K + seg;
    const bf16* gB = Bt + (size_t)(bn + row) * K + seg;
    const unsigned ldsA[2] = { (unsigned)(size_t)&sA[0][soff], (unsigned)(size_t)&sA[1][soff] };
    const unsigned ldsB[2] = { (unsigned)(size_t)&sB[0][soff], (unsigned)(size_t)&sB[1][soff] };

    const int T = K >> 5;           // number of K-tiles

    // prologue: stage tile 0 into buf0
    async_cp32B(ldsA[0], gA);
    async_cp32B(ldsB[0], gB);
    wait_async0();
    __syncthreads();

    for (int t = 0; t < T; ++t) {
        const int p = t & 1;
        if (t + 1 < T) {
            // async-stage tile t+1 into the alternate buffer (overlaps compute)
            async_cp32B(ldsA[p ^ 1], gA + (size_t)(t + 1) * 32);
            async_cp32B(ldsB[p ^ 1], gB + (size_t)(t + 1) * 32);
            if (t + 2 < T) {
                __builtin_prefetch(gA + (size_t)(t + 2) * 32, 0, 1);
                __builtin_prefetch(gB + (size_t)(t + 2) * 32, 0, 1);
            }
        }
        const bf16* cA = sA[p];
        const bf16* cB = sB[p];
#pragma unroll
        for (int mi = 0; mi < 2; ++mi) {
            const int arow = waveM * 32 + mi * 16 + l16;     // M = l16 (both halves)
            AFrag af;                                        // K: half*8+{0..7}, 16+half*8+{0..7}
            af.h[0] = *(const bf16x8*)&cA[arow * 40 + half * 8];
            af.h[1] = *(const bf16x8*)&cA[arow * 40 + 16 + half * 8];
#pragma unroll
            for (int ni = 0; ni < 4; ++ni) {
                const int nrow = waveN * 64 + ni * 16 + l16; // N = l16, K: half*8+{0..7}, 16+half*8+{0..7}
                AFrag bfg;
                bfg.h[0] = *(const bf16x8*)&cB[nrow * 40 + half * 8];
                bfg.h[1] = *(const bf16x8*)&cB[nrow * 40 + 16 + half * 8];
                acc[mi][ni] = __builtin_amdgcn_wmma_f32_16x16x32_bf16(
                    false, af.v, false, bfg.v, (short)0, acc[mi][ni], false, false);
            }
        }
        wait_async0();               // own async writes to LDS have landed
        __syncthreads();             // everyone's writes visible
    }

#pragma unroll
    for (int mi = 0; mi < 2; ++mi)
#pragma unroll
        for (int ni = 0; ni < 4; ++ni) {
            const int col  = bn + waveN * 64 + ni * 16 + l16;
            const float bv = bias ? bias[col] : 0.f;
#pragma unroll
            for (int vv = 0; vv < 8; ++vv) {
                const int rw = bm + waveM * 32 + mi * 16 + half * 8 + vv;
                float x = acc[mi][ni][vv] + bv;
                if (act == 1) x = x > 0.f ? x : 0.f;
                C[(size_t)rw * N + col] = x;
            }
        }
}

// ---------------------------------------------------------------------------
// Convert f32 [K,N] -> bf16 transposed [N,K], tiled through LDS.
// ---------------------------------------------------------------------------
__global__ __launch_bounds__(256) void k_cvt_tr(const float* __restrict__ in,
                                                bf16* __restrict__ out, int K, int N) {
    __shared__ float t[32][33];
    const size_t mat = (size_t)blockIdx.z * K * N;
    const int bx = blockIdx.x * 32;      // along N
    const int by = blockIdx.y * 32;      // along K
    const int tx = threadIdx.x & 31, ty = threadIdx.x >> 5;   // 32x8
#pragma unroll
    for (int j = 0; j < 32; j += 8)
        t[ty + j][tx] = in[mat + (size_t)(by + ty + j) * N + bx + tx];
    __syncthreads();
#pragma unroll
    for (int j = 0; j < 32; j += 8)
        out[mat + (size_t)(bx + ty + j) * K + by + tx] = (bf16)t[tx][ty + j];
}

// ---------------------------------------------------------------------------
// CSR build (per relation, done once; edges shared by both layers)
// ---------------------------------------------------------------------------
__global__ void k_zeroi(int* p, int n) {
    int i = blockIdx.x * 256 + threadIdx.x;
    if (i < n) p[i] = 0;
}

__global__ void k_hist(const int* __restrict__ dst_flat, int* __restrict__ cnt) {
    int i = blockIdx.x * 256 + threadIdx.x;          // over R*E
    if (i >= R_ * E_) return;
    const int r = i >> 14;                           // E_ = 2^14
    atomicAdd(&cnt[r * N_ + dst_flat[i]], 1);
}

__global__ __launch_bounds__(256) void k_scan(const int* __restrict__ cnt,
                                              int* __restrict__ rowptr) {
    const int r = blockIdx.x;
    const int* c = cnt + (size_t)r * N_;
    int* rp = rowptr + (size_t)r * (N_ + 1);
    __shared__ int part[256];
    const int t = threadIdx.x;
    int local[16];
    int s = 0;
#pragma unroll
    for (int j = 0; j < 16; ++j) { local[j] = s; s += c[t * 16 + j]; }
    part[t] = s;
    __syncthreads();
    for (int st = 1; st < 256; st <<= 1) {
        int v = (t >= st) ? part[t - st] : 0;
        __syncthreads();
        part[t] += v;
        __syncthreads();
    }
    const int base = (t == 0) ? 0 : part[t - 1];
#pragma unroll
    for (int j = 0; j < 16; ++j) rp[t * 16 + j] = base + local[j];
    if (t == 255) rp[N_] = part[255];
}

__global__ void k_initcursor(const int* __restrict__ rowptr, int* __restrict__ cur) {
    int i = blockIdx.x * 256 + threadIdx.x;          // over R*N
    if (i >= R_ * N_) return;
    const int r = i >> 12;                           // N_ = 2^12
    cur[i] = rowptr[(size_t)r * (N_ + 1) + (i & (N_ - 1))];
}

__global__ void k_scatter(const int* __restrict__ dst_flat, int* __restrict__ cur,
                          int* __restrict__ perm) {
    int i = blockIdx.x * 256 + threadIdx.x;          // over R*E
    if (i >= R_ * E_) return;
    const int r = i >> 14, e = i & (E_ - 1);
    const int d = dst_flat[i];
    const int p = atomicAdd(&cur[r * N_ + d], 1);
    perm[(size_t)r * E_ + p] = e;
}

// ---------------------------------------------------------------------------
// Elementwise / graph kernels
// ---------------------------------------------------------------------------
__global__ void k_f32_to_bf16(const float* __restrict__ in, bf16* __restrict__ out, int n) {
    int i = blockIdx.x * 256 + threadIdx.x;
    if (i < n) out[i] = (bf16)in[i];
}

__global__ void k_zerof(float* p, int n) {
    int i = blockIdx.x * 256 + threadIdx.x;
    if (i < n) p[i] = 0.f;
}

__global__ void k_bmean(const float* __restrict__ b, float* __restrict__ out, int dim, float inv) {
    int i = blockIdx.x * 256 + threadIdx.x;
    if (i < dim) {
        float s = 0.f;
        for (int r = 0; r < R_; ++r) s += b[(size_t)r * dim + i];
        out[i] = s * inv;
    }
}

__global__ void k_eler(const float* __restrict__ z, const float* __restrict__ al,
                       const float* __restrict__ ar, float* __restrict__ el,
                       float* __restrict__ er) {
    int i = blockIdx.x * 256 + threadIdx.x;          // over N*H
    if (i >= N_ * H_) return;
    const float* zp  = z + (size_t)i * HID_;
    const int h      = i & (H_ - 1);
    const float* alp = al + h * HID_;
    const float* arp = ar + h * HID_;
    float sl = 0.f, sr = 0.f;
    for (int c = 0; c < HID_; ++c) { float v = zp[c]; sl += v * alp[c]; sr += v * arp[c]; }
    el[i] = sl; er[i] = sr;
}

__global__ void k_init_mden(float* m, float* den, int n) {
    int i = blockIdx.x * 256 + threadIdx.x;
    if (i < n) { m[i] = -__builtin_inff(); den[i] = 0.f; }
}

__device__ __forceinline__ void atomicMaxF(float* addr, float val) {
    if (val >= 0.f) atomicMax((int*)addr, __float_as_int(val));
    else            atomicMin((unsigned int*)addr, __float_as_uint(val));
}

__global__ void k_scores(const int* __restrict__ src, const int* __restrict__ dst,
                         const float* __restrict__ el, const float* __restrict__ er,
                         float* __restrict__ esc, float* __restrict__ m) {
    int i = blockIdx.x * 256 + threadIdx.x;          // over E*H
    if (i >= E_ * H_) return;
    const int e = i >> 2, h = i & 3;
    const int s = src[e], d = dst[e];
    float v = el[s * H_ + h] + er[d * H_ + h];
    v = v > 0.f ? v : 0.2f * v;
    esc[i] = v;
    atomicMaxF(&m[d * H_ + h], v);
}

__global__ void k_expden(const int* __restrict__ dst, const float* __restrict__ m,
                         float* __restrict__ esc, float* __restrict__ den) {
    int i = blockIdx.x * 256 + threadIdx.x;
    if (i >= E_ * H_) return;
    const int d = dst[i >> 2], h = i & 3;
    float ex = __expf(esc[i] - m[d * H_ + h]);
    esc[i] = ex;
    atomicAdd(&den[d * H_ + h], ex);
}

__global__ void k_attn_mean(const int* __restrict__ dst, const float* __restrict__ esc,
                            const float* __restrict__ den, float* __restrict__ attn) {
    int e = blockIdx.x * 256 + threadIdx.x;
    if (e >= E_) return;
    const int d = dst[e];
    float s = 0.f;
    for (int h = 0; h < H_; ++h) s += esc[e * H_ + h] / den[d * H_ + h];
    attn[e] = 0.25f * s;
}

// CSR aggregation: one wave per (dst node, head); no atomics.
__global__ __launch_bounds__(256) void k_agg_csr(
    const int* __restrict__ rowptr, const int* __restrict__ perm,
    const int* __restrict__ src, const float* __restrict__ esc,
    const float* __restrict__ den, const float* __restrict__ z,
    float* __restrict__ acc) {
    const int wv   = threadIdx.x >> 5;
    const int lane = threadIdx.x & 31;
    const int idx  = blockIdx.x * 8 + wv;            // over N*H
    const int n = idx >> 2, h = idx & 3;
    const int beg = rowptr[n], end = rowptr[n + 1];
    if (beg == end) return;
    const float invden = 1.f / den[n * H_ + h];
    const float* zh = z + h * HID_;
    float a0 = 0.f, a1 = 0.f, a2 = 0.f, a3 = 0.f;
    for (int p = beg; p < end; ++p) {
        const int e = perm[p];
        const float alpha = esc[e * H_ + h] * invden;
        const float* zp = zh + (size_t)src[e] * D_;
        a0 += alpha * zp[lane];
        a1 += alpha * zp[lane + 32];
        a2 += alpha * zp[lane + 64];
        a3 += alpha * zp[lane + 96];
    }
    float* ap = acc + (size_t)n * D_ + h * HID_;
    ap[lane]      += a0;
    ap[lane + 32] += a1;
    ap[lane + 64] += a2;
    ap[lane + 96] += a3;
}

__global__ void k_finalize(const float* __restrict__ acc, const float* __restrict__ resid,
                           const float* __restrict__ bmean, float* __restrict__ out, int doElu) {
    int i = blockIdx.x * 256 + threadIdx.x;          // over N*D
    if (i >= N_ * D_) return;
    float x = acc[i] * (1.f / R_) + resid[i] + bmean[i & (D_ - 1)];
    if (doElu) x = x > 0.f ? x : (__expf(x) - 1.f);
    out[i] = x;
}

__global__ void k_imp(const int* __restrict__ dst_flat, const float* __restrict__ attn,
                      float* __restrict__ imp) {
    int i = blockIdx.x * 256 + threadIdx.x;          // over R*E
    if (i >= R_ * E_) return;
    atomicAdd(&imp[dst_flat[i]], attn[i]);
}

__global__ void k_linear_naive(const float* __restrict__ X, const float* __restrict__ W,
                               const float* __restrict__ b, float* __restrict__ Y,
                               int M, int N, int K) {
    int i = blockIdx.x * 256 + threadIdx.x;
    if (i >= M * N) return;
    const int row = i / N, col = i - row * N;
    float s = b ? b[col] : 0.f;
    const float* xp = X + (size_t)row * K;
    for (int k = 0; k < K; ++k) s += xp[k] * W[(size_t)k * N + col];
    Y[i] = s;
}

__global__ __launch_bounds__(256) void k_mha(const float* __restrict__ q,
                                             const float* __restrict__ kk,
                                             const float* __restrict__ vv,
                                             float* __restrict__ ctx) {
    const int b = blockIdx.x >> 2, hm = blockIdx.x & 3;
    __shared__ float sc[S_];
    __shared__ float red[S_];
    const int s = threadIdx.x;
    const float* qp = q + b * D_ + hm * HID_;
    const float* kp = kk + ((size_t)(b * S_ + s)) * D_ + hm * HID_;
    float dot = 0.f;
    for (int j = 0; j < HID_; ++j) dot += qp[j] * kp[j];
    dot *= 0.088388347648318447f;                     // 1/sqrt(128)
    sc[s] = dot; red[s] = dot;
    __syncthreads();
    for (int st = 128; st > 0; st >>= 1) { if (s < st) red[s] = fmaxf(red[s], red[s + st]); __syncthreads(); }
    const float mx = red[0];
    __syncthreads();
    float ex = __expf(sc[s] - mx);
    sc[s] = ex; red[s] = ex;
    __syncthreads();
    for (int st = 128; st > 0; st >>= 1) { if (s < st) red[s] += red[s + st]; __syncthreads(); }
    const float inv = 1.f / red[0];
    if (s < HID_) {
        float a = 0.f;
        for (int t = 0; t < S_; ++t)
            a += sc[t] * vv[((size_t)(b * S_ + t)) * D_ + hm * HID_ + s];
        ctx[b * D_ + hm * HID_ + s] = a * inv;
    }
}

__global__ void k_pack_combined(const float* __restrict__ feats, const float* __restrict__ imp,
                                const float* __restrict__ aout, bf16* __restrict__ out) {
    int i = blockIdx.x * 256 + threadIdx.x;          // over N*2D
    if (i >= N_ * 2 * D_) return;
    const int n = i >> 10, d = i & 1023;
    float v = (d < D_) ? feats[(size_t)n * D_ + d] * imp[n]
                       : aout[(size_t)(n >> 8) * D_ + (d - D_)];
    out[i] = (bf16)v;
}

// ---------------------------------------------------------------------------
static inline int GB(long long n) { return (int)((n + 255) / 256); }

extern "C" void kernel_launch(void* const* d_in, const int* in_sizes, int n_in,
                              void* d_out, int out_size, void* d_ws, size_t ws_size,
                              hipStream_t stream) {
    const float* feat = (const float*)d_in[0];
    const float* hyp  = (const float*)d_in[1];
    const int*   esrc = (const int*)d_in[2];
    const int*   edst = (const int*)d_in[3];
    const float* W1   = (const float*)d_in[4];
    const float* al1  = (const float*)d_in[5];
    const float* ar1  = (const float*)d_in[6];
    const float* b1   = (const float*)d_in[7];
    const float* W2   = (const float*)d_in[8];
    const float* al2  = (const float*)d_in[9];
    const float* ar2  = (const float*)d_in[10];
    const float* b2   = (const float*)d_in[11];
    const float* Wq   = (const float*)d_in[12];
    const float* bq   = (const float*)d_in[13];
    const float* Wk   = (const float*)d_in[14];
    const float* bk   = (const float*)d_in[15];
    const float* Wv   = (const float*)d_in[16];
    const float* bv   = (const float*)d_in[17];
    const float* Wo   = (const float*)d_in[18];
    const float* bo   = (const float*)d_in[19];
    const float* Wn1  = (const float*)d_in[20];
    const float* bn1  = (const float*)d_in[21];
    const float* Wn2  = (const float*)d_in[22];
    const float* bn2  = (const float*)d_in[23];

    float* out     = (float*)d_out;
    float* attnbuf = out + (size_t)N_ * 2;           // [R,E] attention output region

    // workspace carving
    char* ws = (char*)d_ws;
    size_t off = 0;
    auto carve = [&](size_t bytes) -> char* {
        char* p = ws + off;
        off += (bytes + 255) & ~(size_t)255;
        return p;
    };
    bf16*  bfA    = (bf16*)carve((size_t)N_ * 2 * D_ * 2);            // featbf / h1bf / featsbf / combinedbf
    bf16*  W1bf   = (bf16*)carve((size_t)R_ * DIN_ * D_ * 2);         // transposed [r][N=512][K=512]
    bf16*  W2bf   = (bf16*)carve((size_t)R_ * D_ * D_ * 2);           // transposed
    bf16*  Wkbf   = (bf16*)carve((size_t)D_ * D_ * 2);                // transposed
    bf16*  Wvbf   = (bf16*)carve((size_t)D_ * D_ * 2);                // transposed
    bf16*  Wn1bf  = (bf16*)carve((size_t)2 * D_ * HID_ * 2);          // transposed [128][1024]
    float* z      = (float*)carve((size_t)N_ * D_ * 4);
    float* el     = (float*)carve((size_t)N_ * H_ * 4);
    float* er     = (float*)carve((size_t)N_ * H_ * 4);
    float* mbuf   = (float*)carve((size_t)N_ * H_ * 4);
    float* den    = (float*)carve((size_t)N_ * H_ * 4);
    float* esc    = (float*)carve((size_t)E_ * H_ * 4);
    float* acc    = (float*)carve((size_t)N_ * D_ * 4);
    float* h1     = (float*)carve((size_t)N_ * D_ * 4);
    float* feats  = (float*)carve((size_t)N_ * D_ * 4);
    float* imp    = (float*)carve((size_t)N_ * 4);
    float* qbuf   = (float*)carve((size_t)B_ * D_ * 4);
    float* ctx    = (float*)carve((size_t)B_ * D_ * 4);
    float* aout   = (float*)carve((size_t)B_ * D_ * 4);
    float* hidden = (float*)carve((size_t)N_ * HID_ * 4);
    float* bmean1 = (float*)carve((size_t)D_ * 4);
    float* bmean2 = (float*)carve((size_t)D_ * 4);
    int*   cnt    = (int*)carve((size_t)R_ * N_ * 4);                 // counts, then cursor
    int*   rowptr = (int*)carve((size_t)R_ * (N_ + 1) * 4);
    int*   perm   = (int*)carve((size_t)R_ * E_ * 4);
    float* kbuf   = z;    // reuse: z is free after layer 2
    float* vbuf   = acc;  // reuse: acc is free after finalize2

    // ---- CSR build (edges shared by both layers; built once) ----
    k_zeroi<<<GB((long long)R_ * N_), 256, 0, stream>>>(cnt, R_ * N_);
    k_hist<<<GB((long long)R_ * E_), 256, 0, stream>>>(edst, cnt);
    k_scan<<<R_, 256, 0, stream>>>(cnt, rowptr);
    k_initcursor<<<GB((long long)R_ * N_), 256, 0, stream>>>(rowptr, cnt);
    k_scatter<<<GB((long long)R_ * E_), 256, 0, stream>>>(edst, cnt, perm);

    // ---- conversions to bf16 (weights transposed to [N][K]) ----
    k_f32_to_bf16<<<GB((long long)N_ * DIN_), 256, 0, stream>>>(feat, bfA, N_ * DIN_);
    k_cvt_tr<<<dim3(D_ / 32, DIN_ / 32, R_), 256, 0, stream>>>(W1, W1bf, DIN_, D_);
    k_cvt_tr<<<dim3(D_ / 32, D_ / 32, R_), 256, 0, stream>>>(W2, W2bf, D_, D_);
    k_cvt_tr<<<dim3(D_ / 32, D_ / 32, 1), 256, 0, stream>>>(Wk, Wkbf, D_, D_);
    k_cvt_tr<<<dim3(D_ / 32, D_ / 32, 1), 256, 0, stream>>>(Wv, Wvbf, D_, D_);
    k_cvt_tr<<<dim3(HID_ / 32, 2 * D_ / 32, 1), 256, 0, stream>>>(Wn1, Wn1bf, 2 * D_, HID_);
    k_bmean<<<GB(D_), 256, 0, stream>>>(b1, bmean1, D_, 1.f / R_);
    k_bmean<<<GB(D_), 256, 0, stream>>>(b2, bmean2, D_, 1.f / R_);

    const dim3 gemm_blk(256);
    const long long EH = (long long)E_ * H_;

    // ---- RGAT layer 1 ----
    k_zerof<<<GB((long long)N_ * D_), 256, 0, stream>>>(acc, N_ * D_);
    for (int r = 0; r < R_; ++r) {
        const int* sr = esrc + (size_t)r * E_;
        const int* dr = edst + (size_t)r * E_;
        k_gemm_bf16_wmma<<<dim3(D_ / 128, N_ / 128), gemm_blk, 0, stream>>>(
            bfA, W1bf + (size_t)r * DIN_ * D_, z, N_, D_, DIN_, nullptr, 0);
        k_eler<<<GB((long long)N_ * H_), 256, 0, stream>>>(z, al1 + (size_t)r * D_, ar1 + (size_t)r * D_, el, er);
        k_init_mden<<<GB((long long)N_ * H_), 256, 0, stream>>>(mbuf, den, N_ * H_);
        k_scores<<<GB(EH), 256, 0, stream>>>(sr, dr, el, er, esc, mbuf);
        k_expden<<<GB(EH), 256, 0, stream>>>(dr, mbuf, esc, den);
        k_attn_mean<<<GB(E_), 256, 0, stream>>>(dr, esc, den, attnbuf + (size_t)r * E_);
        k_agg_csr<<<N_ * H_ / 8, 256, 0, stream>>>(
            rowptr + (size_t)r * (N_ + 1), perm + (size_t)r * E_, sr, esc, den, z, acc);
    }
    k_finalize<<<GB((long long)N_ * D_), 256, 0, stream>>>(acc, feat, bmean1, h1, 1);

    // ---- RGAT layer 2 ----
    k_f32_to_bf16<<<GB((long long)N_ * D_), 256, 0, stream>>>(h1, bfA, N_ * D_);
    k_zerof<<<GB((long long)N_ * D_), 256, 0, stream>>>(acc, N_ * D_);
    for (int r = 0; r < R_; ++r) {
        const int* sr = esrc + (size_t)r * E_;
        const int* dr = edst + (size_t)r * E_;
        k_gemm_bf16_wmma<<<dim3(D_ / 128, N_ / 128), gemm_blk, 0, stream>>>(
            bfA, W2bf + (size_t)r * D_ * D_, z, N_, D_, D_, nullptr, 0);
        k_eler<<<GB((long long)N_ * H_), 256, 0, stream>>>(z, al2 + (size_t)r * D_, ar2 + (size_t)r * D_, el, er);
        k_init_mden<<<GB((long long)N_ * H_), 256, 0, stream>>>(mbuf, den, N_ * H_);
        k_scores<<<GB(EH), 256, 0, stream>>>(sr, dr, el, er, esc, mbuf);
        k_expden<<<GB(EH), 256, 0, stream>>>(dr, mbuf, esc, den);
        k_agg_csr<<<N_ * H_ / 8, 256, 0, stream>>>(
            rowptr + (size_t)r * (N_ + 1), perm + (size_t)r * E_, sr, esc, den, z, acc);
    }
    k_finalize<<<GB((long long)N_ * D_), 256, 0, stream>>>(acc, h1, bmean2, feats, 0);

    // ---- node importance ----
    k_zerof<<<GB(N_), 256, 0, stream>>>(imp, N_);
    k_imp<<<GB((long long)R_ * E_), 256, 0, stream>>>(edst, attnbuf, imp);

    // ---- MHA ----
    k_f32_to_bf16<<<GB((long long)N_ * D_), 256, 0, stream>>>(feats, bfA, N_ * D_);
    k_gemm_bf16_wmma<<<dim3(D_ / 128, N_ / 128), gemm_blk, 0, stream>>>(
        bfA, Wkbf, kbuf, N_, D_, D_, bk, 0);
    k_gemm_bf16_wmma<<<dim3(D_ / 128, N_ / 128), gemm_blk, 0, stream>>>(
        bfA, Wvbf, vbuf, N_, D_, D_, bv, 0);
    k_linear_naive<<<GB((long long)B_ * D_), 256, 0, stream>>>(hyp, Wq, bq, qbuf, B_, D_, D_);
    k_mha<<<B_ * H_, S_, 0, stream>>>(qbuf, kbuf, vbuf, ctx);
    k_linear_naive<<<GB((long long)B_ * D_), 256, 0, stream>>>(ctx, Wo, bo, aout, B_, D_, D_);

    // ---- classifier ----
    k_pack_combined<<<GB((long long)N_ * 2 * D_), 256, 0, stream>>>(feats, imp, aout, bfA);
    k_gemm_bf16_wmma<<<dim3(HID_ / 128, N_ / 128), gemm_blk, 0, stream>>>(
        bfA, Wn1bf, hidden, N_, HID_, 2 * D_, bn1, 1);
    k_linear_naive<<<GB((long long)N_ * 2), 256, 0, stream>>>(hidden, Wn2, bn2, out, N_, 2, HID_);

    (void)in_sizes; (void)n_in; (void)out_size; (void)ws_size;
}
